// MultiHeadAttention_28544352649302
// MI455X (gfx1250) — matrix operand order repaired and checked by reading, not compiled, not run on hardware
//
#include <hip/hip_runtime.h>

#define DEV __device__ __forceinline__

typedef _Float16 v16h __attribute__((ext_vector_type(16)));
typedef float    v8f  __attribute__((ext_vector_type(8)));
typedef unsigned int u32x4 __attribute__((ext_vector_type(4)));
typedef int      i32x4 __attribute__((ext_vector_type(4)));
typedef int      i32x8 __attribute__((ext_vector_type(8)));

static constexpr int Bb = 2;
static constexpr int Ss = 2048;
static constexpr int Dm = 1024;
static constexpr int Hh = 16;
static constexpr int Dk = 64;

// ---- Tensor Data Mover (CDNA5): guarded so missing builtins fall back to
// plain cooperative uint4 copies. ----
#if defined(__AMDGCN__) && __has_builtin(__builtin_amdgcn_tensor_load_to_lds) && \
    __has_builtin(__builtin_amdgcn_s_wait_tensorcnt)
#define HAVE_TDM 1
#else
#define HAVE_TDM 0
#endif

#if HAVE_TDM
// 2D tile load global -> LDS via TDM. tile0 = contiguous elements per row,
// tile1 = rows, stride0 = row stride in elements (f16, data_size=2B).
// D# layout per CDNA5 ISA.
DEV void tdm_load_2d(const void* gsrc, unsigned lds_addr,
                     unsigned tile0, unsigned tile1, unsigned long long stride0) {
  unsigned long long va = (unsigned long long)(uintptr_t)gsrc;
  u32x4 g0;
  g0[0] = 1u;                                               // count=1 (valid), user mode
  g0[1] = lds_addr;                                         // lds_addr [63:32]
  g0[2] = (unsigned)va;                                     // global_addr lo
  g0[3] = (unsigned)((va >> 32) & 0x01FFFFFFull) | 0x80000000u;  // global_addr hi | type=2
  unsigned tdim0 = tile0, tdim1 = tile1;
  unsigned long long s0 = stride0, s1 = stride0;
  i32x8 g1;
  g1[0] = (int)(1u << 16);                                  // data_size=1 (2 bytes), no mask
  g1[1] = (int)((tdim0 & 0xFFFFu) << 16);                   // tensor_dim0[15:0] @ bits 63:48
  g1[2] = (int)(((tdim0 >> 16) & 0xFFFFu) | ((tdim1 & 0xFFFFu) << 16));
  g1[3] = (int)(((tdim1 >> 16) & 0xFFFFu) | ((tile0 & 0xFFFFu) << 16));  // | tile_dim0
  g1[4] = (int)(tile1 & 0xFFFFu);                           // tile_dim1 ; tile_dim2=0
  g1[5] = (int)(unsigned)(s0 & 0xFFFFFFFFull);              // tensor_dim0_stride lo
  g1[6] = (int)(unsigned)(((s0 >> 32) & 0xFFFFull) | ((s1 & 0xFFFFull) << 16));
  g1[7] = (int)(unsigned)((s1 >> 16) & 0xFFFFFFFFull);      // tensor_dim1_stride hi
  i32x4 z4 = {};
#if __clang_major__ >= 23
  i32x8 z8 = {};
  __builtin_amdgcn_tensor_load_to_lds(g0, g1, z4, z4, z8, 0);
#else
  __builtin_amdgcn_tensor_load_to_lds(g0, g1, z4, z4, 0);
#endif
}
#endif

// ---- WMMA fragment helpers (wave32; A and B both use k = 16*(lane/16) + e) ----
DEV int a_k_idx(int lane, int e) {
  return ((lane & 16) ? 16 : 0) + e;
}

DEV v16h load_a_frag(const _Float16* rowp, int lane) {
  v16h a;
#pragma unroll
  for (int e = 0; e < 16; ++e) a[e] = rowp[a_k_idx(lane, e)];
  return a;
}

// B fragment from LDS tile stored [n][k]
DEV v16h load_b_frag_nk(const _Float16* tile, int ld, int lane, int nbase, int kbase) {
  const _Float16* p = tile + (size_t)(nbase + (lane & 15)) * ld + kbase + ((lane & 16) ? 16 : 0);
  v16h b;
#pragma unroll
  for (int e = 0; e < 16; ++e) b[e] = p[e];
  return b;
}

// B fragment from LDS tile stored [k][n]
DEV v16h load_b_frag_kn(const _Float16* tile, int ld, int lane, int nbase, int kbase) {
  int n  = nbase + (lane & 15);
  int k0 = kbase + ((lane & 16) ? 16 : 0);
  v16h b;
#pragma unroll
  for (int e = 0; e < 16; ++e) b[e] = tile[(size_t)(k0 + e) * ld + n];
  return b;
}

DEV v8f wmma_f16(v16h a, v16h b, v8f c) {
  return __builtin_amdgcn_wmma_f32_16x16x32_f16(false, a, false, b, (short)0, c, false, false);
}

// ---- fp32 -> f16 conversion ----
__global__ void cvt_f32_to_f16(const float* __restrict__ in, _Float16* __restrict__ out, int n) {
  int i = blockIdx.x * blockDim.x + threadIdx.x;
  int stride = gridDim.x * blockDim.x;
  for (; i < n; i += stride) out[i] = (_Float16)in[i];
}

// ---- Tiled WMMA GEMM: C[M,N] = A[M,K] * W[N,K]^T + bias ----
// 64x64 tile per WG, 128 threads (4 waves), K-step 64, double-buffered TDM staging.
template <bool HEAD, typename OutT>
__global__ __launch_bounds__(128)
void gemm_wmma_kernel(const _Float16* __restrict__ A, const _Float16* __restrict__ W,
                      const float* __restrict__ bias, OutT* __restrict__ out,
                      int M, int N, int K) {
  __shared__ __attribute__((aligned(16))) _Float16 At[2][64][64];
  __shared__ __attribute__((aligned(16))) _Float16 Bt[2][64][64];
  const int tid  = threadIdx.x;
  const int lane = tid & 31;
  const int wv   = tid >> 5;
  const int m0   = blockIdx.x * 64;
  const int n0   = blockIdx.y * 64;

  v8f acc[4] = {};

#if HAVE_TDM
  const unsigned ldsA0 = (unsigned)(uintptr_t)&At[0][0][0];
  const unsigned ldsA1 = (unsigned)(uintptr_t)&At[1][0][0];
  const unsigned ldsB0 = (unsigned)(uintptr_t)&Bt[0][0][0];
  const unsigned ldsB1 = (unsigned)(uintptr_t)&Bt[1][0][0];
  if (wv == 0) {  // prologue: stage k0 = 0 into buffer 0
    tdm_load_2d(&A[(size_t)m0 * K], ldsA0, 64, 64, (unsigned long long)K);
    tdm_load_2d(&W[(size_t)n0 * K], ldsB0, 64, 64, (unsigned long long)K);
  }
#endif

  for (int k0 = 0; k0 < K; k0 += 64) {
    const int cur = (k0 >> 6) & 1;
    __syncthreads();  // nobody still reads the buffer we are about to fill
#if HAVE_TDM
    if (wv == 0) {
      if (k0 + 64 < K) {  // prefetch next K-step into the alternate buffer
        unsigned la = cur ? ldsA0 : ldsA1;
        unsigned lb = cur ? ldsB0 : ldsB1;
        tdm_load_2d(&A[(size_t)m0 * K + k0 + 64], la, 64, 64, (unsigned long long)K);
        tdm_load_2d(&W[(size_t)n0 * K + k0 + 64], lb, 64, 64, (unsigned long long)K);
        __builtin_amdgcn_s_wait_tensorcnt(2);  // in-order: current pair has landed
      } else {
        __builtin_amdgcn_s_wait_tensorcnt(0);
      }
    }
#else
    for (int idx = tid; idx < 512; idx += 128) {
      int row = idx >> 3, c = (idx & 7) * 8;
      *(uint4*)(&At[cur][row][c]) = *(const uint4*)(&A[(size_t)(m0 + row) * K + k0 + c]);
      *(uint4*)(&Bt[cur][row][c]) = *(const uint4*)(&W[(size_t)(n0 + row) * K + k0 + c]);
    }
#endif
    __syncthreads();

    v16h a0 = load_a_frag(&At[cur][wv * 16 + (lane & 15)][0], lane);
    v16h a1 = load_a_frag(&At[cur][wv * 16 + (lane & 15)][32], lane);
    v16h bfr[8];
#pragma unroll
    for (int f = 0; f < 4; ++f) {
      bfr[f * 2]     = load_b_frag_nk(&Bt[cur][0][0], 64, lane, f * 16, 0);
      bfr[f * 2 + 1] = load_b_frag_nk(&Bt[cur][0][0], 64, lane, f * 16, 32);
    }
#pragma unroll
    for (int f = 0; f < 4; ++f) {
      acc[f] = wmma_f16(a0, bfr[f * 2], acc[f]);
      acc[f] = wmma_f16(a1, bfr[f * 2 + 1], acc[f]);
    }
  }

  const int mrow = (lane & 16) ? 8 : 0;
#pragma unroll
  for (int f = 0; f < 4; ++f) {
    int n = n0 + f * 16 + (lane & 15);
    float bv = bias[n];
#pragma unroll
    for (int r = 0; r < 8; ++r) {
      int m = m0 + wv * 16 + r + mrow;
      float val = acc[f][r] + bv;
      if (HEAD) {
        int b = m / Ss, s = m % Ss;
        int h = n >> 6, d = n & 63;
        ((_Float16*)out)[(((size_t)b * Hh + h) * Ss + s) * Dk + d] = (_Float16)val;
      } else {
        ((float*)out)[(size_t)m * N + n] = val;
      }
    }
  }
}

// ---- Flash-style attention: one WG per (b, h, 64 q-rows), 4 waves ----
__global__ __launch_bounds__(128)
void attention_kernel(const _Float16* __restrict__ Qg, const _Float16* __restrict__ Kg,
                      const _Float16* __restrict__ Vg, const int* __restrict__ mask,
                      _Float16* __restrict__ Ao) {
  __shared__ __attribute__((aligned(16))) _Float16 Qt[64][64];
  __shared__ __attribute__((aligned(16))) _Float16 Kt[2][64][64];
  __shared__ __attribute__((aligned(16))) _Float16 Vt[2][64][64];
  __shared__ __attribute__((aligned(16))) _Float16 Pt[64][64];

  const int tid  = threadIdx.x;
  const int lane = tid & 31;
  const int wv   = tid >> 5;
  const int q0   = blockIdx.x * 64;
  const int h    = blockIdx.y;
  const int b    = blockIdx.z;

  const _Float16* Qbase = Qg + (((size_t)b * Hh + h) * Ss + q0) * Dk;
  const _Float16* Kbase = Kg + (((size_t)b * Hh + h) * Ss) * Dk;
  const _Float16* Vbase = Vg + (((size_t)b * Hh + h) * Ss) * Dk;
  const int* mbase = mask + (size_t)b * Ss * Ss;
  const int NJ = Ss / 64;

#if HAVE_TDM
  const unsigned ldsQ  = (unsigned)(uintptr_t)&Qt[0][0];
  const unsigned ldsK0 = (unsigned)(uintptr_t)&Kt[0][0][0];
  const unsigned ldsK1 = (unsigned)(uintptr_t)&Kt[1][0][0];
  const unsigned ldsV0 = (unsigned)(uintptr_t)&Vt[0][0][0];
  const unsigned ldsV1 = (unsigned)(uintptr_t)&Vt[1][0][0];
  if (wv == 0) {  // pipeline: Q, then K0/V0 (tiles are contiguous 4096 halves)
    tdm_load_2d(Qbase, ldsQ, 4096, 1, 4096ull);
    tdm_load_2d(Kbase, ldsK0, 4096, 1, 4096ull);
    tdm_load_2d(Vbase, ldsV0, 4096, 1, 4096ull);
    __builtin_amdgcn_s_wait_tensorcnt(2);  // in-order: Q has landed
  }
#else
  {
    const uint4* src = (const uint4*)Qbase;
    uint4* dst = (uint4*)&Qt[0][0];
    for (int i = tid; i < 512; i += 128) dst[i] = src[i];
  }
#endif
  __syncthreads();

  v16h aq0 = load_a_frag(&Qt[wv * 16 + (lane & 15)][0], lane);
  v16h aq1 = load_a_frag(&Qt[wv * 16 + (lane & 15)][32], lane);

  v8f o[4] = {};
  float m_st[8], l_st[8];
#pragma unroll
  for (int r = 0; r < 8; ++r) { m_st[r] = -1e30f; l_st[r] = 0.0f; }

  const int mrow = (lane & 16) ? 8 : 0;
  const float scale = 0.125f;  // 1/sqrt(64)

  for (int j = 0; j < NJ; ++j) {
    const int cur = j & 1;
    __syncthreads();  // nobody still reads the buffer about to be refilled
#if HAVE_TDM
    if (wv == 0) {
      if (j + 1 < NJ) {  // prefetch next key block into alternate buffer
        unsigned lk = cur ? ldsK0 : ldsK1;
        unsigned lv = cur ? ldsV0 : ldsV1;
        tdm_load_2d(Kbase + (size_t)(j + 1) * 64 * Dk, lk, 4096, 1, 4096ull);
        tdm_load_2d(Vbase + (size_t)(j + 1) * 64 * Dk, lv, 4096, 1, 4096ull);
        __builtin_amdgcn_s_wait_tensorcnt(2);  // in-order: block j has landed
      } else {
        __builtin_amdgcn_s_wait_tensorcnt(0);
      }
    }
#else
    {
      const uint4* ks = (const uint4*)(Kbase + (size_t)j * 64 * Dk);
      const uint4* vs = (const uint4*)(Vbase + (size_t)j * 64 * Dk);
      uint4* kd = (uint4*)&Kt[cur][0][0];
      uint4* vd = (uint4*)&Vt[cur][0][0];
      for (int i = tid; i < 512; i += 128) { kd[i] = ks[i]; vd[i] = vs[i]; }
    }
#endif
    __syncthreads();

    // scores S = Q * K^T (reduction over d = 64); batch all B frags first
    v16h bk[8];
#pragma unroll
    for (int f = 0; f < 4; ++f) {
      bk[f * 2]     = load_b_frag_nk(&Kt[cur][0][0], 64, lane, f * 16, 0);
      bk[f * 2 + 1] = load_b_frag_nk(&Kt[cur][0][0], 64, lane, f * 16, 32);
    }
    v8f s[4];
#pragma unroll
    for (int f = 0; f < 4; ++f) {
      v8f z = {};
      z = wmma_f16(aq0, bk[f * 2], z);
      z = wmma_f16(aq1, bk[f * 2 + 1], z);
      s[f] = z;
    }

    // scale + mask (mask==0 -> -1e9)
#pragma unroll
    for (int f = 0; f < 4; ++f) {
      int t = j * 64 + f * 16 + (lane & 15);
#pragma unroll
      for (int r = 0; r < 8; ++r) {
        int q = q0 + wv * 16 + r + mrow;
        int mk = mbase[(size_t)q * Ss + t];
        s[f][r] = mk ? s[f][r] * scale : -1e9f;
      }
    }

    // online softmax; row m lives in register slot r across a 16-lane half-wave
#pragma unroll
    for (int r = 0; r < 8; ++r) {
      float rm = fmaxf(fmaxf(s[0][r], s[1][r]), fmaxf(s[2][r], s[3][r]));
#pragma unroll
      for (int off = 8; off >= 1; off >>= 1) rm = fmaxf(rm, __shfl_xor(rm, off, 32));
      float mnew  = fmaxf(m_st[r], rm);
      float alpha = __expf(m_st[r] - mnew);
      float rs = 0.0f;
#pragma unroll
      for (int f = 0; f < 4; ++f) {
        float p = __expf(s[f][r] - mnew);
        s[f][r] = p;
        rs += p;
      }
#pragma unroll
      for (int off = 8; off >= 1; off >>= 1) rs += __shfl_xor(rs, off, 32);
      l_st[r] = l_st[r] * alpha + rs;
      m_st[r] = mnew;
#pragma unroll
      for (int f = 0; f < 4; ++f) o[f][r] *= alpha;
    }

    // P: C-fragment layout -> LDS -> A-fragment layout (f16)
#pragma unroll
    for (int f = 0; f < 4; ++f)
#pragma unroll
      for (int r = 0; r < 8; ++r)
        Pt[wv * 16 + r + mrow][f * 16 + (lane & 15)] = (_Float16)s[f][r];
    __syncthreads();

    v16h ap0 = load_a_frag(&Pt[wv * 16 + (lane & 15)][0], lane);
    v16h ap1 = load_a_frag(&Pt[wv * 16 + (lane & 15)][32], lane);
    v16h bvf[8];
#pragma unroll
    for (int f = 0; f < 4; ++f) {
      bvf[f * 2]     = load_b_frag_kn(&Vt[cur][0][0], 64, lane, f * 16, 0);
      bvf[f * 2 + 1] = load_b_frag_kn(&Vt[cur][0][0], 64, lane, f * 16, 32);
    }
#pragma unroll
    for (int f = 0; f < 4; ++f) {
      o[f] = wmma_f16(ap0, bvf[f * 2], o[f]);
      o[f] = wmma_f16(ap1, bvf[f * 2 + 1], o[f]);
    }
  }

  // normalize and write merged-head layout [B,S,Dm] (f16) for the O-projection
#pragma unroll
  for (int f = 0; f < 4; ++f) {
#pragma unroll
    for (int r = 0; r < 8; ++r) {
      int q = q0 + wv * 16 + r + mrow;
      int d = h * Dk + f * 16 + (lane & 15);
      float val = o[f][r] / l_st[r];
      Ao[((size_t)b * Ss + q) * Dm + d] = (_Float16)val;
    }
  }
}

extern "C" void kernel_launch(void* const* d_in, const int* in_sizes, int n_in,
                              void* d_out, int out_size, void* d_ws, size_t ws_size,
                              hipStream_t stream) {
  (void)in_sizes; (void)n_in; (void)out_size; (void)ws_size;

  const float* x  = (const float*)d_in[0];
  const int* mask = (const int*)d_in[1];
  const float* Wq = (const float*)d_in[2];
  const float* bq = (const float*)d_in[3];
  const float* Wk = (const float*)d_in[4];
  const float* bk = (const float*)d_in[5];
  const float* Wv = (const float*)d_in[6];
  const float* bv = (const float*)d_in[7];
  const float* Wo = (const float*)d_in[8];
  const float* bo = (const float*)d_in[9];
  float* out = (float*)d_out;

  char* ws = (char*)d_ws;
  const size_t MB = 1024 * 1024;
  _Float16* xh  = (_Float16*)(ws + 0);        // 8 MB  [B,S,D] f16
  _Float16* Wqh = (_Float16*)(ws + 8  * MB);  // 2 MB
  _Float16* Wkh = (_Float16*)(ws + 10 * MB);
  _Float16* Wvh = (_Float16*)(ws + 12 * MB);
  _Float16* Woh = (_Float16*)(ws + 14 * MB);
  _Float16* Qh  = (_Float16*)(ws + 16 * MB);  // 8 MB  [B,H,S,Dk] f16
  _Float16* Kh  = (_Float16*)(ws + 24 * MB);
  _Float16* Vh  = (_Float16*)(ws + 32 * MB);
  _Float16* Ah  = (_Float16*)(ws + 40 * MB);  // 8 MB  [B,S,D] f16

  const int NX = Bb * Ss * Dm;  // 4,194,304
  const int NW = Dm * Dm;       // 1,048,576
  cvt_f32_to_f16<<<1024, 256, 0, stream>>>(x,  xh,  NX);
  cvt_f32_to_f16<<<512,  256, 0, stream>>>(Wq, Wqh, NW);
  cvt_f32_to_f16<<<512,  256, 0, stream>>>(Wk, Wkh, NW);
  cvt_f32_to_f16<<<512,  256, 0, stream>>>(Wv, Wvh, NW);
  cvt_f32_to_f16<<<512,  256, 0, stream>>>(Wo, Woh, NW);

  dim3 gg(Bb * Ss / 64, Dm / 64);  // (64,16)
  gemm_wmma_kernel<true, _Float16><<<gg, 128, 0, stream>>>(xh, Wqh, bq, Qh, Bb * Ss, Dm, Dm);
  gemm_wmma_kernel<true, _Float16><<<gg, 128, 0, stream>>>(xh, Wkh, bk, Kh, Bb * Ss, Dm, Dm);
  gemm_wmma_kernel<true, _Float16><<<gg, 128, 0, stream>>>(xh, Wvh, bv, Vh, Bb * Ss, Dm, Dm);

  dim3 ga(Ss / 64, Hh, Bb);  // (32,16,2)
  attention_kernel<<<ga, 128, 0, stream>>>(Qh, Kh, Vh, mask, Ah);

  gemm_wmma_kernel<false, float><<<gg, 128, 0, stream>>>(Ah, Woh, bo, out, Bb * Ss, Dm, Dm);
}
